// GcnAttentionCell_46385646797645
// MI455X (gfx1250) — matrix operand order repaired and checked by reading, not compiled, not run on hardware
//
#include <hip/hip_runtime.h>

#define B_ 64
#define N_ 207
#define T_ 24
#define D_ 128
#define H_ 8
#define M_ (B_ * N_ * T_)              // 317952, divisible by 128
#define MD_ ((size_t)M_ * (size_t)D_)  // 40,697,856 elements

typedef __attribute__((ext_vector_type(16))) _Float16 v16h;
typedef __attribute__((ext_vector_type(8)))  float    v8f;
typedef unsigned int uint;

// ---------------------------------------------------------------------------
// CDNA5 async global->LDS copy (16B granule), tracked by ASYNCcnt.
// Generic pointers to __shared__ carry the LDS byte offset in their low 32
// bits (ISA 10.2), so truncation gives the ds offset.
// ---------------------------------------------------------------------------
__device__ __forceinline__ void async_ld16(uint lds_byte, uint voff, const void* sbase)
{
    asm volatile("global_load_async_to_lds_b128 %0, %1, %2"
                 :: "v"(lds_byte), "v"(voff), "s"(sbase) : "memory");
}
__device__ __forceinline__ void async_wait0()
{
    asm volatile("s_wait_asynccnt 0x0" ::: "memory");
}

// swizzled LDS layout for 64B rows of 32 f16: granule' = granule ^ ((row>>2)&3)
__device__ __forceinline__ uint swz_off(int row, int k)  // element (row, k) byte offset
{
    return (uint)(row * 64 + ((((k >> 3) ^ (row >> 2)) & 3) * 16) + (k & 7) * 2);
}

// ---------------------------------------------------------------------------
// OUT[M][128] = X1 @ W1^T (+ X2 @ W2^T) + bias.  X*, W* are f16.
// W element layout: W[j * wpitch + k].  Output f16 (out16) or f32 (out32).
// One block = 128 rows, 8 waves; wave w -> row-tile w x 8 col-tiles.
// Staging is pure async byte copy into XOR-swizzled LDS.
// ---------------------------------------------------------------------------
__global__ __launch_bounds__(256) void gemm128_kernel(
    const _Float16* __restrict__ X1, const _Float16* __restrict__ W1,
    const _Float16* __restrict__ X2, const _Float16* __restrict__ W2,
    int wpitch, const float* __restrict__ bias,
    float* __restrict__ out32, _Float16* __restrict__ out16, int nsrc)
{
    __shared__ _Float16 sX[128 * 32];   // [row][k] swizzled, 64B rows
    __shared__ _Float16 sW[128 * 32];   // [outch][k] swizzled

    const int tid  = threadIdx.x;
    const int wave = tid >> 5;
    const int lane = tid & 31;
    const int lh   = lane >> 4;
    const int ln   = lane & 15;
    const int r0   = blockIdx.x * 128;
    const uint xBase = (uint)(uintptr_t)&sX[0];
    const uint wBase = (uint)(uintptr_t)&sW[0];

    v8f acc[8] = {};

    const int nchunks = nsrc * 4;
    for (int c = 0; c < nchunks; ++c) {
        const _Float16* X = (c < 4) ? X1 : X2;
        const _Float16* W = (c < 4) ? W1 : W2;
        const int kc = (c & 3) * 32;
        const char* xs = (const char*)X + (size_t)r0 * 256 + kc * 2;  // X rowpitch 128 f16
        const char* ws = (const char*)W + kc * 2;

        // 512 granules per array, 2 per thread: row = e>>2, granule g = e&3
        #pragma unroll
        for (int i = 0; i < 2; ++i) {
            int e   = tid + 256 * i;
            int row = e >> 2;
            int g   = e & 3;
            uint dst = (uint)(row * 64 + (((g ^ (row >> 2)) & 3) * 16));
            async_ld16(xBase + dst, (uint)(row * 256 + g * 16), xs);
            async_ld16(wBase + dst, (uint)(row * (wpitch * 2) + g * 16), ws);
        }
        async_wait0();
        __syncthreads();

        // A fragment (16x32): lane ln = row m, K halves by lh
        union { uint u[8]; v16h v; } ua;
        const int m = wave * 16 + ln;
        #pragma unroll
        for (int vv = 0; vv < 8; ++vv) {
            int kb = (vv < 4 ? 2 * vv : 16 + 2 * (vv - 4)) + 8 * lh;
            ua.u[vv] = *(const uint*)&sX[swz_off(m, kb) >> 1];
        }
        #pragma unroll
        for (int ct = 0; ct < 8; ++ct) {
            // B fragment (32x16): lane ln = column j, K halves by lh (same order as A)
            union { uint u[8]; v16h v; } ub;
            const int j  = ct * 16 + ln;
            const int ks = 16 * lh;
            #pragma unroll
            for (int vv = 0; vv < 8; ++vv) {
                ub.u[vv] = *(const uint*)&sW[swz_off(j, (vv < 4 ? 2 * vv : 16 + 2 * (vv - 4)) + 8 * lh) >> 1];
            }
            acc[ct] = __builtin_amdgcn_wmma_f32_16x16x32_f16(
                false, ua.v, false, ub.v, (short)0, acc[ct], false, false);
        }
        __syncthreads();
    }

    if (out16) {
        #pragma unroll
        for (int ct = 0; ct < 8; ++ct)
            #pragma unroll
            for (int r = 0; r < 8; ++r) {
                int row = r0 + wave * 16 + r + 8 * lh;
                int col = ct * 16 + ln;
                out16[(size_t)row * 128 + col] = (_Float16)(acc[ct][r] + bias[col]);
            }
    } else {
        #pragma unroll
        for (int ct = 0; ct < 8; ++ct)
            #pragma unroll
            for (int r = 0; r < 8; ++r) {
                int row = r0 + wave * 16 + r + 8 * lh;
                int col = ct * 16 + ln;
                out32[(size_t)row * 128 + col] = acc[ct][r] + bias[col];
            }
    }
}

// ---------------------------------------------------------------------------
// GCN aggregation: per (b,t)  agg[m][c] = sum_n matrix[b,t,n,m] * h[b,n,t,c]
// A = matrix^T staged manually (transpose + zero-pad, f32 source);
// B = hidden slice async-staged from f16.  Wave w owns col-tile w, B-frag
// reused across all 13 row tiles.
// ---------------------------------------------------------------------------
__global__ __launch_bounds__(256) void agg_kernel(
    const _Float16* __restrict__ h16, const float* __restrict__ matrix,
    _Float16* __restrict__ agg16)
{
    __shared__ _Float16 sA[208 * 34];   // A[m][kk], pitch 34 (conflict-free)
    __shared__ _Float16 sB[32 * 128];   // B[kk][c], 256B contiguous rows

    const int tid  = threadIdx.x;
    const int wave = tid >> 5;
    const int lane = tid & 31;
    const int lh   = lane >> 4;
    const int ln   = lane & 15;
    const int b    = blockIdx.x / T_;
    const int t    = blockIdx.x % T_;
    const float* Mt = matrix + ((size_t)b * T_ + t) * N_ * N_;
    const uint bBase = (uint)(uintptr_t)&sB[0];

    v8f acc[13] = {};

    for (int kb = 0; kb < 7; ++kb) {           // K = n, 7 chunks of 32 (pad 224)
        const int kc = kb * 32;
        // stage A[m][kk] = Mt[kc+kk][m] (transpose), zero-padded
        for (int e = tid; e < 32 * 208; e += 256) {
            int kk = e / 208;
            int m  = e % 208;
            float val = 0.0f;
            if (kc + kk < N_ && m < N_) val = Mt[(size_t)(kc + kk) * N_ + m];
            sA[m * 34 + kk] = (_Float16)val;
        }
        // stage B[kk][c] = h16[b, kc+kk, t, c] via async copy; zero invalid rows
        {
            int nvalid = N_ - kc; if (nvalid > 32) nvalid = 32;
            const char* hb = (const char*)h16 + (((size_t)b * N_ + kc) * T_ + t) * 256;
            #pragma unroll
            for (int i = 0; i < 2; ++i) {
                int e  = tid + 256 * i;        // 512 granules: 32 rows x 16
                int kk = e >> 4;
                int g  = e & 15;
                if (kk < nvalid) {
                    async_ld16(bBase + (uint)(kk * 256 + g * 16),
                               (uint)(kk * (T_ * 256) + g * 16), hb);
                } else {
                    *(uint4*)&sB[kk * 128 + g * 8] = make_uint4(0u, 0u, 0u, 0u);
                }
            }
        }
        async_wait0();
        __syncthreads();

        // B fragment for this wave's column tile (element reads, k-major rows)
        v16h bfrag;
        {
            const int col = wave * 16 + ln;
            const int ks  = 16 * lh;
            #pragma unroll
            for (int vv = 0; vv < 8; ++vv) {
                bfrag[2 * vv]     = sB[((vv < 4 ? 2 * vv : 16 + 2 * (vv - 4)) + 8 * lh)     * 128 + col];
                bfrag[2 * vv + 1] = sB[((vv < 4 ? 2 * vv : 16 + 2 * (vv - 4)) + 8 * lh + 1) * 128 + col];
            }
        }
        #pragma unroll
        for (int rt = 0; rt < 13; ++rt) {
            v16h a;
            const int m = rt * 16 + ln;
            #pragma unroll
            for (int vv = 0; vv < 8; ++vv) {
                int kk = (vv < 4 ? 2 * vv : 16 + 2 * (vv - 4)) + 8 * lh;
                a[2 * vv]     = sA[m * 34 + kk];
                a[2 * vv + 1] = sA[m * 34 + kk + 1];
            }
            acc[rt] = __builtin_amdgcn_wmma_f32_16x16x32_f16(
                false, a, false, bfrag, (short)0, acc[rt], false, false);
        }
        __syncthreads();
    }

    for (int rt = 0; rt < 13; ++rt) {
        #pragma unroll
        for (int r = 0; r < 8; ++r) {
            int m   = rt * 16 + r + 8 * lh;
            int col = wave * 16 + ln;
            if (m < N_)
                agg16[(((size_t)b * N_ + m) * T_ + t) * 128 + col] = (_Float16)acc[rt][r];
        }
    }
}

// ---------------------------------------------------------------------------
// Causal MHA per (b,n): T=24, H=8, DK=16.  f16 slabs async-staged to LDS,
// f32 math, one thread per (h,t).
// ---------------------------------------------------------------------------
__global__ __launch_bounds__(256) void attn_kernel(
    const _Float16* __restrict__ q, const _Float16* __restrict__ k,
    const _Float16* __restrict__ v, _Float16* __restrict__ ctx)
{
    __shared__ _Float16 sq[3072], sk[3072], sv[3072];
    const int tid = threadIdx.x;
    const size_t base = (size_t)blockIdx.x * T_ * 128;   // elements
    const uint qB = (uint)(uintptr_t)&sq[0];
    const uint kB = (uint)(uintptr_t)&sk[0];
    const uint vB = (uint)(uintptr_t)&sv[0];
    // 3072 f16 = 384 granules per array
    for (int e = tid; e < 384; e += 256) {
        uint off = (uint)e * 16;
        async_ld16(qB + off, off, (const char*)q + base * 2);
        async_ld16(kB + off, off, (const char*)k + base * 2);
        async_ld16(vB + off, off, (const char*)v + base * 2);
    }
    async_wait0();
    __syncthreads();

    if (tid < 192) {
        const int h = tid & 7;
        const int t = tid >> 3;
        const _Float16* qp = &sq[t * 128 + h * 16];
        const float scale = 0.25f;                       // 1/sqrt(16)
        float s[24];
        float mmax = -1e30f;
        #pragma unroll
        for (int j = 0; j < 24; ++j) {
            const _Float16* kp = &sk[j * 128 + h * 16];
            float d = 0.f;
            #pragma unroll
            for (int e = 0; e < 16; ++e) d += (float)qp[e] * (float)kp[e];
            s[j] = (j <= t) ? d * scale : -1e30f;
            mmax = fmaxf(mmax, s[j]);
        }
        float ssum = 0.f;
        #pragma unroll
        for (int j = 0; j < 24; ++j) { s[j] = __expf(s[j] - mmax); ssum += s[j]; }
        const float inv = 1.f / ssum;
        float o[16];
        #pragma unroll
        for (int e = 0; e < 16; ++e) o[e] = 0.f;
        #pragma unroll
        for (int j = 0; j < 24; ++j) {
            const _Float16* vp = &sv[j * 128 + h * 16];
            float p = s[j] * inv;
            #pragma unroll
            for (int e = 0; e < 16; ++e) o[e] += p * (float)vp[e];
        }
        _Float16* op = ctx + base + t * 128 + h * 16;
        #pragma unroll
        for (int e = 0; e < 16; ++e) op[e] = (_Float16)o[e];
    }
}

// ---------------------------------------------------------------------------
// f32 -> f16 conversion (vectorized), n4 = element count / 4
// ---------------------------------------------------------------------------
__global__ __launch_bounds__(256) void conv_kernel(
    const float* __restrict__ src, _Float16* __restrict__ dst, long n4)
{
    const size_t stride = (size_t)gridDim.x * 256;
    for (size_t i = (size_t)blockIdx.x * 256 + threadIdx.x; i < (size_t)n4; i += stride) {
        float4 f = ((const float4*)src)[i];
        union { _Float16 h[4]; uint2 u; } cv;
        cv.h[0] = (_Float16)f.x; cv.h[1] = (_Float16)f.y;
        cv.h[2] = (_Float16)f.z; cv.h[3] = (_Float16)f.w;
        ((uint2*)dst)[i] = cv.u;
    }
}

// ---------------------------------------------------------------------------
// BatchNorm stats over (B,N,T): stats[0..127]=sum, [128..255]=sumsq
// ---------------------------------------------------------------------------
__global__ __launch_bounds__(256) void stats_kernel(
    const float* __restrict__ g, float* __restrict__ stats)
{
    __shared__ float ssum[256], ssq[256];
    const int tid = threadIdx.x;
    const size_t stride = (size_t)gridDim.x * 256;        // multiple of 128
    float a = 0.f, bsq = 0.f;
    for (size_t i = (size_t)blockIdx.x * 256 + tid; i < MD_; i += stride) {
        float x = g[i];
        a += x; bsq += x * x;
    }
    ssum[tid] = a; ssq[tid] = bsq;
    __syncthreads();
    if (tid < 128) {
        atomicAdd(&stats[tid],       ssum[tid] + ssum[tid + 128]);
        atomicAdd(&stats[128 + tid], ssq[tid]  + ssq[tid + 128]);
    }
}

// ---------------------------------------------------------------------------
// z = sigmoid(BN(g)); out = z*gcn + (1-z)*attn
// ---------------------------------------------------------------------------
__global__ __launch_bounds__(256) void fuse_kernel(
    const float* __restrict__ g, const _Float16* __restrict__ gcn16,
    const _Float16* __restrict__ attn16, const float* __restrict__ gamma,
    const float* __restrict__ beta, const float* __restrict__ stats,
    float* __restrict__ out)
{
    const size_t stride = (size_t)gridDim.x * 256;
    const float invM = 1.0f / (float)M_;
    for (size_t i = (size_t)blockIdx.x * 256 + threadIdx.x; i < MD_; i += stride) {
        int ch = (int)(i & 127);
        float mean = stats[ch] * invM;
        float var  = stats[128 + ch] * invM - mean * mean;
        float gh = (g[i] - mean) * rsqrtf(var + 1e-5f) * gamma[ch] + beta[ch];
        float z  = 1.f / (1.f + __expf(-gh));
        out[i] = z * (float)gcn16[i] + (1.f - z) * (float)attn16[i];
    }
}

// ---------------------------------------------------------------------------
extern "C" void kernel_launch(void* const* d_in, const int* in_sizes, int n_in,
                              void* d_out, int out_size, void* d_ws, size_t ws_size,
                              hipStream_t stream)
{
    const float* hidden = (const float*)d_in[0];
    const float* matrix = (const float*)d_in[1];
    const float* Wq     = (const float*)d_in[2];
    const float* bq     = (const float*)d_in[3];
    const float* Wk     = (const float*)d_in[4];
    const float* bk     = (const float*)d_in[5];
    const float* Wv     = (const float*)d_in[6];
    const float* bv     = (const float*)d_in[7];
    const float* Wo     = (const float*)d_in[8];
    const float* bo     = (const float*)d_in[9];
    const float* Wgcn   = (const float*)d_in[10];
    const float* bgcn   = (const float*)d_in[11];
    const float* Wgate  = (const float*)d_in[12];
    const float* bgate  = (const float*)d_in[13];
    const float* gamma  = (const float*)d_in[14];
    const float* beta   = (const float*)d_in[15];
    float* out = (float*)d_out;

    _Float16* h16   = (_Float16*)d_ws;
    _Float16* q16   = h16 + MD_;     // later reused as attnX (f16 attn_out)
    _Float16* k16   = q16 + MD_;     // later reused as aggX
    _Float16* v16   = k16 + MD_;     // later reused as gcnX (f16 gcn_out)
    _Float16* ctx16 = v16 + MD_;
    float* g        = (float*)(ctx16 + MD_);
    float* stats    = g + MD_;
    _Float16* wq16  = (_Float16*)(stats + 256);
    _Float16* wk16   = wq16 + 16384;
    _Float16* wv16   = wk16 + 16384;
    _Float16* wo16   = wv16 + 16384;
    _Float16* wgcn16 = wo16 + 16384;
    _Float16* wgate16 = wgcn16 + 16384;   // 128 x 256

    const int GB = M_ / 128;              // 2484 row blocks
    dim3 blk(256);

    conv_kernel<<<2048, blk, 0, stream>>>(hidden, h16, (long)(MD_ / 4));
    conv_kernel<<<16, blk, 0, stream>>>(Wq,   wq16,   4096);
    conv_kernel<<<16, blk, 0, stream>>>(Wk,   wk16,   4096);
    conv_kernel<<<16, blk, 0, stream>>>(Wv,   wv16,   4096);
    conv_kernel<<<16, blk, 0, stream>>>(Wo,   wo16,   4096);
    conv_kernel<<<16, blk, 0, stream>>>(Wgcn, wgcn16, 4096);
    conv_kernel<<<32, blk, 0, stream>>>(Wgate, wgate16, 8192);

    gemm128_kernel<<<GB, blk, 0, stream>>>(h16, wq16, nullptr, nullptr, 128, bq, nullptr, q16, 1);
    gemm128_kernel<<<GB, blk, 0, stream>>>(h16, wk16, nullptr, nullptr, 128, bk, nullptr, k16, 1);
    gemm128_kernel<<<GB, blk, 0, stream>>>(h16, wv16, nullptr, nullptr, 128, bv, nullptr, v16, 1);
    attn_kernel<<<B_ * N_, blk, 0, stream>>>(q16, k16, v16, ctx16);
    gemm128_kernel<<<GB, blk, 0, stream>>>(ctx16, wo16, nullptr, nullptr, 128, bo, nullptr, q16, 1);  // attnX
    agg_kernel<<<B_ * T_, blk, 0, stream>>>(h16, matrix, k16);                                        // aggX
    gemm128_kernel<<<GB, blk, 0, stream>>>(k16, wgcn16, nullptr, nullptr, 128, bgcn, nullptr, v16, 1); // gcnX
    hipMemsetAsync(stats, 0, 256 * sizeof(float), stream);
    gemm128_kernel<<<GB, blk, 0, stream>>>(v16, wgate16, q16, wgate16 + 128, 256, bgate, g, nullptr, 2); // g
    stats_kernel<<<1024, blk, 0, stream>>>(g, stats);
    fuse_kernel<<<2048, blk, 0, stream>>>(g, v16, q16, gamma, beta, stats, out);
}
